// Mamba2EnhancedBlock_61211873902954
// MI455X (gfx1250) — compile-verified, matrix operand order checked by reading, not run on hardware
//
#include <hip/hip_runtime.h>
#include <hip/hip_bf16.h>
#include <math.h>

// ---------------------------------------------------------------------------
// Mamba2EnhancedBlock forward for MI455X (gfx1250, wave32, WMMA).
// Fragment-major LDS staging: all WMMA operands are single 32B LDS reads.
// ---------------------------------------------------------------------------

typedef __attribute__((ext_vector_type(16))) _Float16 v16h;
typedef __attribute__((ext_vector_type(8)))  _Float16 v8h;
typedef __attribute__((ext_vector_type(8)))  float    v8f;

#define B_   4
#define L_   1024
#define DM_  512
#define DIN_ 1024
#define NHh_ 16          // mamba heads
#define HD_  64          // mamba headdim
#define DST_ 64          // d_state
#define DIP_ 2192        // d_in_proj
#define CVD_ 1152        // conv dim
#define AH_  8           // attention heads
#define ADH_ 64          // attention head dim
#define ROWS_ (B_ * L_)  // 4096

__device__ __forceinline__ float siluf(float x) { return x / (1.f + __expf(-x)); }

// Convert two float4 (8 consecutive values) to packed 8 x f16, with scale.
__device__ __forceinline__ v8h cvt8(float4 a, float4 b, float s) {
  v8h h;
  h[0] = (_Float16)(a.x * s); h[1] = (_Float16)(a.y * s);
  h[2] = (_Float16)(a.z * s); h[3] = (_Float16)(a.w * s);
  h[4] = (_Float16)(b.x * s); h[5] = (_Float16)(b.y * s);
  h[6] = (_Float16)(b.z * s); h[7] = (_Float16)(b.w * s);
  return h;
}

// ---------------------------------------------------------------------------
// LayerNorm over D=512, one wave per row, float4 I/O.
// ---------------------------------------------------------------------------
__global__ __launch_bounds__(256) void ln512_kernel(
    const float* __restrict__ in, const float* __restrict__ g,
    const float* __restrict__ b, float* __restrict__ out) {
  int wave = threadIdx.x >> 5, lane = threadIdx.x & 31;
  int row = blockIdx.x * 8 + wave;
  const float4* xv = (const float4*)(in + (size_t)row * DM_) + lane * 4;
  float4 f[4];
  float s = 0.f;
#pragma unroll
  for (int i = 0; i < 4; ++i) {
    f[i] = xv[i];
    s += f[i].x + f[i].y + f[i].z + f[i].w;
  }
#pragma unroll
  for (int m = 16; m >= 1; m >>= 1) s += __shfl_xor(s, m, 32);
  float mu = s * (1.f / DM_);
  float v = 0.f;
#pragma unroll
  for (int i = 0; i < 4; ++i) {
    float dx = f[i].x - mu, dy = f[i].y - mu, dz = f[i].z - mu, dw = f[i].w - mu;
    v += dx * dx + dy * dy + dz * dz + dw * dw;
  }
#pragma unroll
  for (int m = 16; m >= 1; m >>= 1) v += __shfl_xor(v, m, 32);
  float rs = rsqrtf(v * (1.f / DM_) + 1e-5f);
  const float4* gv = (const float4*)g + lane * 4;
  const float4* bv = (const float4*)b + lane * 4;
  float4* ov = (float4*)(out + (size_t)row * DM_) + lane * 4;
#pragma unroll
  for (int i = 0; i < 4; ++i) {
    float4 gg = gv[i], bb = bv[i], o;
    o.x = (f[i].x - mu) * rs * gg.x + bb.x;
    o.y = (f[i].y - mu) * rs * gg.y + bb.y;
    o.z = (f[i].z - mu) * rs * gg.z + bb.z;
    o.w = (f[i].w - mu) * rs * gg.w + bb.w;
    ov[i] = o;
  }
}

// ---------------------------------------------------------------------------
// GEMM: C[M,N] = A[M,K] @ W[K,N] + bias (+ residual).
// 128 thr = 4 waves; block tile 64x64, K-step 32; fragment-major LDS.
// OOB columns handled by address clamp + select (no divergent loads).
// ---------------------------------------------------------------------------
__global__ __launch_bounds__(128) void wmma_gemm_kernel(
    const float* __restrict__ A, const float* __restrict__ W,
    const float* __restrict__ bias, const float* __restrict__ res,
    float* __restrict__ C, int M, int N, int K) {
  __shared__ __align__(32) _Float16 Af[4][32][16];  // [rowblk][lane][e]
  __shared__ __align__(32) _Float16 Wf[4][32][16];  // [colblk][lane][e]
  int tid = threadIdx.x;
  int lane = tid & 31, wave = tid >> 5;
  int half = lane >> 4, mn = lane & 15;
  int bm0 = blockIdx.y * 64, bn0 = blockIdx.x * 64;
  v8f acc[4];
#pragma unroll
  for (int t = 0; t < 4; ++t) acc[t] = (v8f){};

  for (int k0 = 0; k0 < K; k0 += 32) {
    __syncthreads();
    // A tile 64x32: task p -> row r = p>>2, k-octet g = p&3.
#pragma unroll
    for (int pi = 0; pi < 2; ++pi) {
      int p = tid + pi * 128;
      int r = p >> 2, g = p & 3;
      const float* src = A + (size_t)(bm0 + r) * K + k0 + g * 8;
      float4 f0 = ((const float4*)src)[0];
      float4 f1 = ((const float4*)src)[1];
      *(v8h*)&Af[r >> 4][(r & 15) + 16 * (g & 1)][8 * (g >> 1)] =
          cvt8(f0, f1, 1.f);
    }
    // W tile 32x64: task p -> col n = p&63, k-octet g = p>>6.
    // Clamp column (always in-bounds), issue all 8 loads first (one clause),
    // then convert + zero-select.
#pragma unroll
    for (int pi = 0; pi < 2; ++pi) {
      int p = tid + pi * 128;
      int n = p & 63, g = p >> 6;
      int col = bn0 + n;
      bool valid = col < N;
      int colc = valid ? col : (N - 1);
      const float* src = W + (size_t)(k0 + g * 8) * N + colc;
      float tmp[8];
#pragma unroll
      for (int j = 0; j < 8; ++j) tmp[j] = src[(size_t)j * N];
      v8h h;
#pragma unroll
      for (int j = 0; j < 8; ++j)
        h[j] = valid ? (_Float16)tmp[j] : (_Float16)0.f;
      *(v8h*)&Wf[n >> 4][(n & 15) + 16 * (g & 1)][8 * (g >> 1)] = h;
    }
    __syncthreads();
    // Prefetch next K-step tiles into cache (global_prefetch_b8).
    if (k0 + 32 < K) {
      __builtin_prefetch(A + (size_t)(bm0 + (tid >> 1)) * K + k0 + 32 +
                             (tid & 1) * 16, 0, 1);
      __builtin_prefetch(W + (size_t)(k0 + 32 + (tid >> 4)) * N + bn0 +
                             (tid & 15) * 4, 0, 1);
    }
    v16h a = *(const v16h*)Af[wave][lane];
#pragma unroll
    for (int t = 0; t < 4; ++t) {
      v16h bb = *(const v16h*)Wf[t][lane];
      acc[t] = __builtin_amdgcn_wmma_f32_16x16x32_f16(false, a, false, bb,
                                                      (short)0, acc[t],
                                                      false, false);
    }
  }
#pragma unroll
  for (int t = 0; t < 4; ++t)
#pragma unroll
    for (int i = 0; i < 8; ++i) {
      int r = bm0 + wave * 16 + half * 8 + i;
      int c = bn0 + t * 16 + mn;
      if (c < N) {
        float v = acc[t][i] + bias[c];
        if (res) v += res[(size_t)r * N + c];
        C[(size_t)r * N + c] = v;
      }
    }
}

// ---------------------------------------------------------------------------
// Depthwise causal conv (width 4) + SiLU; softplus dt.
// zxbcdt row: [z:0..1023 | xBC:1024..2175 | dt:2176..2191]
// ---------------------------------------------------------------------------
__global__ __launch_bounds__(256) void conv_dt_kernel(
    const float* __restrict__ big, const float* __restrict__ conv_w,
    const float* __restrict__ conv_b, const float* __restrict__ dt_bias,
    float* __restrict__ xbc, float* __restrict__ dt) {
  int row = blockIdx.x;  // b*L + l
  int l = row & (L_ - 1);
  for (int c = threadIdx.x; c < CVD_; c += 256) {
    float acc = conv_b[c];
#pragma unroll
    for (int k = 0; k < 4; ++k) {
      int lp = l + k - 3;
      if (lp >= 0)
        acc += big[(size_t)(row + k - 3) * DIP_ + DIN_ + c] * conv_w[c * 4 + k];
    }
    xbc[(size_t)row * CVD_ + c] = siluf(acc);
  }
  if (threadIdx.x < NHh_) {
    int j = threadIdx.x;
    float t = big[(size_t)row * DIP_ + (DIP_ - NHh_) + j] + dt_bias[j];
    dt[row * NHh_ + j] = (t > 20.f) ? t : log1pf(__expf(t));
  }
}

// ---------------------------------------------------------------------------
// Selective scan: one block per (b, mamba-head); 64x64 state in registers.
// ---------------------------------------------------------------------------
__global__ __launch_bounds__(256) void scan_kernel(
    const float* __restrict__ xbc, const float* __restrict__ dt,
    const float* __restrict__ A_log, const float* __restrict__ D_skip,
    float* __restrict__ y) {
  int b = blockIdx.x >> 4, h = blockIdx.x & 15;
  int tid = threadIdx.x;
  int p = tid >> 2, nq = (tid & 3) * 16;
  __shared__ float xs[HD_], Bs[DST_], Cs[DST_];
  __shared__ float sdA, sdt;
  float aexp = __expf(A_log[h]);
  float dsk = D_skip[h];
  float hs[16];
#pragma unroll
  for (int i = 0; i < 16; ++i) hs[i] = 0.f;
  for (int t = 0; t < L_; ++t) {
    size_t rb = (size_t)(b * L_ + t) * CVD_;
    if (tid < 64)        xs[tid]       = xbc[rb + h * HD_ + tid];
    else if (tid < 128)  Bs[tid - 64]  = xbc[rb + DIN_ + (tid - 64)];
    else if (tid < 192)  Cs[tid - 128] = xbc[rb + DIN_ + DST_ + (tid - 128)];
    else if (tid == 192) {
      float dv = dt[(b * L_ + t) * NHh_ + h];
      sdt = dv;
      sdA = __expf(-aexp * dv);
    }
    __syncthreads();
    float dx = sdt * xs[p];
    float dA = sdA;
    float part = 0.f;
#pragma unroll
    for (int j = 0; j < 16; ++j) {
      int n = nq + j;
      hs[j] = hs[j] * dA + dx * Bs[n];
      part += hs[j] * Cs[n];
    }
    part += __shfl_xor(part, 1, 32);
    part += __shfl_xor(part, 2, 32);
    if ((tid & 3) == 0)
      y[(size_t)(b * L_ + t) * DIN_ + h * HD_ + p] = part + dsk * xs[p];
    __syncthreads();
  }
}

// ---------------------------------------------------------------------------
// Gated RMSNorm (float4): y = y*silu(z); y *= rsqrt(mean(y^2)+eps) * gnorm.
// ---------------------------------------------------------------------------
__global__ __launch_bounds__(256) void gated_norm_kernel(
    float* __restrict__ y, const float* __restrict__ big,
    const float* __restrict__ gn) {
  int row = blockIdx.x, tid = threadIdx.x;
  __shared__ float red[8];
  float4 z4 = ((const float4*)(big + (size_t)row * DIP_))[tid];
  float4 y4 = ((float4*)(y + (size_t)row * DIN_))[tid];
  float4 v;
  v.x = y4.x * siluf(z4.x);
  v.y = y4.y * siluf(z4.y);
  v.z = y4.z * siluf(z4.z);
  v.w = y4.w * siluf(z4.w);
  float s = v.x * v.x + v.y * v.y + v.z * v.z + v.w * v.w;
#pragma unroll
  for (int m = 16; m >= 1; m >>= 1) s += __shfl_xor(s, m, 32);
  if ((tid & 31) == 0) red[tid >> 5] = s;
  __syncthreads();
  float tot = 0.f;
#pragma unroll
  for (int i = 0; i < 8; ++i) tot += red[i];
  float rs = rsqrtf(tot * (1.f / DIN_) + 1e-5f);
  float4 g4 = ((const float4*)gn)[tid];
  float4 o;
  o.x = v.x * rs * g4.x; o.y = v.y * rs * g4.y;
  o.z = v.z * rs * g4.z; o.w = v.w * rs * g4.w;
  ((float4*)(y + (size_t)row * DIN_))[tid] = o;
}

// ---------------------------------------------------------------------------
// Flash attention: block per (b, head, 64-row Q tile), 4 waves.
// All WMMA operands staged fragment-major; Q fragments live in registers.
// ---------------------------------------------------------------------------
__global__ __launch_bounds__(128) void flash_attn_kernel(
    const float* __restrict__ qkv, float* __restrict__ o) {
  int bh = blockIdx.y;
  int b = bh >> 3, hd = bh & 7;
  int q0 = blockIdx.x * 64;
  int tid = threadIdx.x, lane = tid & 31, wave = tid >> 5;
  int half = lane >> 4, mn = lane & 15;

  __shared__ __align__(32) _Float16 Qf[4][2][32][16];  // [rowblk][kblk][lane][e]
  __shared__ __align__(32) _Float16 Kf[2][2][32][16];  // [keyblk][kblk][lane][e]
  __shared__ __align__(32) _Float16 Vf[4][32][16];     // [dblk][lane][e]
  __shared__ __align__(32) _Float16 Pf[4][32][16];     // [wave][lane][e]

  // Stage Q (scaled by 1/sqrt(64)); task p -> row r = p>>3, d-octet g = p&7.
#pragma unroll
  for (int pi = 0; pi < 4; ++pi) {
    int p = tid + pi * 128;
    int r = p >> 3, g = p & 7;
    const float* src =
        qkv + (size_t)(b * L_ + q0 + r) * (3 * DM_) + hd * ADH_ + g * 8;
    float4 f0 = ((const float4*)src)[0];
    float4 f1 = ((const float4*)src)[1];
    *(v8h*)&Qf[r >> 4][g >> 2][(r & 15) + 16 * (g & 1)][8 * ((g >> 1) & 1)] =
        cvt8(f0, f1, 0.125f);
  }
  __syncthreads();
  v16h aq0 = *(const v16h*)Qf[wave][0][lane];
  v16h aq1 = *(const v16h*)Qf[wave][1][lane];

  float m_i[8], l_i[8];
  v8f acc[4];
#pragma unroll
  for (int i = 0; i < 8; ++i) { m_i[i] = -3.0e38f; l_i[i] = 0.f; }
#pragma unroll
  for (int t = 0; t < 4; ++t) acc[t] = (v8f){};

  for (int j = 0; j < L_ / 32; ++j) {
    int k0 = j * 32;
    __syncthreads();
    // K tile 32x64 -> B-fragment layout (contract over d).
#pragma unroll
    for (int pi = 0; pi < 2; ++pi) {
      int p = tid + pi * 128;
      int r = p >> 3, g = p & 7;
      const float* src =
          qkv + (size_t)(b * L_ + k0 + r) * (3 * DM_) + DM_ + hd * ADH_ + g * 8;
      float4 f0 = ((const float4*)src)[0];
      float4 f1 = ((const float4*)src)[1];
      *(v8h*)&Kf[r >> 4][g >> 2][(r & 15) + 16 * (g & 1)][8 * ((g >> 1) & 1)] =
          cvt8(f0, f1, 1.f);
    }
    // V tile 32x64 -> B-fragment layout (contract over key).
    // Issue all 8 strided loads first (one clause), then convert.
#pragma unroll
    for (int pi = 0; pi < 2; ++pi) {
      int p = tid + pi * 128;
      int n = p & 63, g = p >> 6;
      const float* src = qkv + (size_t)(b * L_ + k0 + g * 8) * (3 * DM_) +
                         2 * DM_ + hd * ADH_ + n;
      float tmp[8];
#pragma unroll
      for (int jj = 0; jj < 8; ++jj) tmp[jj] = src[(size_t)jj * (3 * DM_)];
      v8h h;
#pragma unroll
      for (int jj = 0; jj < 8; ++jj) h[jj] = (_Float16)tmp[jj];
      *(v8h*)&Vf[n >> 4][(n & 15) + 16 * (g & 1)][8 * (g >> 1)] = h;
    }
    __syncthreads();

    // S = Q.K^T  (16 rows x 32 keys per wave)
    v8f s0 = {}, s1 = {};
    {
      v16h b00 = *(const v16h*)Kf[0][0][lane];
      s0 = __builtin_amdgcn_wmma_f32_16x16x32_f16(false, aq0, false, b00,
                                                  (short)0, s0, false, false);
      v16h b01 = *(const v16h*)Kf[0][1][lane];
      s0 = __builtin_amdgcn_wmma_f32_16x16x32_f16(false, aq1, false, b01,
                                                  (short)0, s0, false, false);
      v16h b10 = *(const v16h*)Kf[1][0][lane];
      s1 = __builtin_amdgcn_wmma_f32_16x16x32_f16(false, aq0, false, b10,
                                                  (short)0, s1, false, false);
      v16h b11 = *(const v16h*)Kf[1][1][lane];
      s1 = __builtin_amdgcn_wmma_f32_16x16x32_f16(false, aq1, false, b11,
                                                  (short)0, s1, false, false);
    }

    // Online softmax; write exp(P) straight into A-fragment positions.
#pragma unroll
    for (int i = 0; i < 8; ++i) {
      float mx = fmaxf(s0[i], s1[i]);
#pragma unroll
      for (int m = 8; m >= 1; m >>= 1) mx = fmaxf(mx, __shfl_xor(mx, m, 32));
      float mnew = fmaxf(m_i[i], mx);
      float p0 = __expf(s0[i] - mnew);
      float p1 = __expf(s1[i] - mnew);
      float rsum = p0 + p1;
#pragma unroll
      for (int m = 8; m >= 1; m >>= 1) rsum += __shfl_xor(rsum, m, 32);
      float alpha = __expf(m_i[i] - mnew);
      l_i[i] = l_i[i] * alpha + rsum;
      m_i[i] = mnew;
#pragma unroll
      for (int t = 0; t < 4; ++t) acc[t][i] *= alpha;
      int dl = (half * 8 + i) + 16 * ((mn >> 3) & 1);
      Pf[wave][dl][mn & 7] = (_Float16)p0;        // key = mn
      Pf[wave][dl][(mn & 7) + 8] = (_Float16)p1;  // key = 16+mn
    }

    // O += P.V  (per-wave LDS patch; per-wave DS ordering suffices)
    v16h ap = *(const v16h*)Pf[wave][lane];
#pragma unroll
    for (int t = 0; t < 4; ++t) {
      v16h bv = *(const v16h*)Vf[t][lane];
      acc[t] = __builtin_amdgcn_wmma_f32_16x16x32_f16(false, ap, false, bv,
                                                      (short)0, acc[t],
                                                      false, false);
    }
  }

#pragma unroll
  for (int t = 0; t < 4; ++t)
#pragma unroll
    for (int i = 0; i < 8; ++i) {
      int r = q0 + wave * 16 + half * 8 + i;
      o[(size_t)(b * L_ + r) * DM_ + hd * ADH_ + t * 16 + mn] =
          acc[t][i] / l_i[i];
    }
}

// ---------------------------------------------------------------------------
// g = gelu_exact(ff[:, :1024]) * ff[:, 1024:2048]   (float4)
// ---------------------------------------------------------------------------
__global__ __launch_bounds__(256) void gelu_mul_kernel(
    const float* __restrict__ ff, float* __restrict__ g) {
  int idx = blockIdx.x * 256 + threadIdx.x;  // over ROWS_*256 float4s
  int row = idx >> 8, c4 = idx & 255;
  float4 x1 = ((const float4*)(ff + (size_t)row * 2048))[c4];
  float4 x2 = ((const float4*)(ff + (size_t)row * 2048 + 1024))[c4];
  const float k = 0.70710678118654752f;
  float4 o;
  o.x = 0.5f * x1.x * (1.f + erff(x1.x * k)) * x2.x;
  o.y = 0.5f * x1.y * (1.f + erff(x1.y * k)) * x2.y;
  o.z = 0.5f * x1.z * (1.f + erff(x1.z * k)) * x2.z;
  o.w = 0.5f * x1.w * (1.f + erff(x1.w * k)) * x2.w;
  ((float4*)g)[idx] = o;
}

// ---------------------------------------------------------------------------
// Launch
// ---------------------------------------------------------------------------
extern "C" void kernel_launch(void* const* d_in, const int* in_sizes, int n_in,
                              void* d_out, int out_size, void* d_ws,
                              size_t ws_size, hipStream_t stream) {
  (void)in_sizes; (void)n_in; (void)out_size; (void)ws_size;
  const float* x       = (const float*)d_in[0];
  const float* g1      = (const float*)d_in[1];
  const float* b1      = (const float*)d_in[2];
  const float* W_in    = (const float*)d_in[3];
  const float* b_in    = (const float*)d_in[4];
  const float* conv_w  = (const float*)d_in[5];
  const float* conv_b  = (const float*)d_in[6];
  const float* dt_bias = (const float*)d_in[7];
  const float* A_log   = (const float*)d_in[8];
  const float* D_skip  = (const float*)d_in[9];
  const float* gnorm   = (const float*)d_in[10];
  const float* W_out   = (const float*)d_in[11];
  const float* b_out   = (const float*)d_in[12];
  const float* ga      = (const float*)d_in[13];
  const float* ba      = (const float*)d_in[14];
  const float* Wqkv    = (const float*)d_in[15];
  const float* bqkv    = (const float*)d_in[16];
  const float* Wo      = (const float*)d_in[17];
  const float* bo      = (const float*)d_in[18];
  const float* g2      = (const float*)d_in[19];
  const float* b2      = (const float*)d_in[20];
  const float* W1      = (const float*)d_in[21];
  const float* b1f     = (const float*)d_in[22];
  const float* W2      = (const float*)d_in[23];
  const float* b2f     = (const float*)d_in[24];
  float* out = (float*)d_out;

  // Workspace (floats), ~85 MB.
  float* ws  = (float*)d_ws;
  float* h   = ws;                           // 4096*512
  float* big = h   + (size_t)ROWS_ * DM_;    // 4096*2192 (zxbcdt/qkv/ff)
  float* xbc = big + (size_t)ROWS_ * DIP_;   // 4096*1152 (conv out / gate)
  float* dtb = xbc + (size_t)ROWS_ * CVD_;   // 4096*16
  float* yb  = dtb + (size_t)ROWS_ * NHh_;   // 4096*1024 (scan y / attn o)
  float* x2  = yb  + (size_t)ROWS_ * DIN_;   // 4096*512  (residual stream)

  dim3 blk128(128), blk256(256);

  // ---- SSM branch ----
  ln512_kernel<<<ROWS_ / 8, blk256, 0, stream>>>(x, g1, b1, h);
  {
    dim3 grid((DIP_ + 63) / 64, ROWS_ / 64);
    wmma_gemm_kernel<<<grid, blk128, 0, stream>>>(h, W_in, b_in, nullptr, big,
                                                  ROWS_, DIP_, DM_);
  }
  conv_dt_kernel<<<ROWS_, blk256, 0, stream>>>(big, conv_w, conv_b, dt_bias,
                                               xbc, dtb);
  scan_kernel<<<B_ * NHh_, blk256, 0, stream>>>(xbc, dtb, A_log, D_skip, yb);
  gated_norm_kernel<<<ROWS_, blk256, 0, stream>>>(yb, big, gnorm);
  {  // x2 = x + y @ W_out + b_out
    dim3 grid(DM_ / 64, ROWS_ / 64);
    wmma_gemm_kernel<<<grid, blk128, 0, stream>>>(yb, W_out, b_out, x, x2,
                                                  ROWS_, DM_, DIN_);
  }

  // ---- Attention branch ----
  ln512_kernel<<<ROWS_ / 8, blk256, 0, stream>>>(x2, ga, ba, h);
  {  // qkv = h @ Wqkv + bqkv
    dim3 grid((3 * DM_) / 64, ROWS_ / 64);
    wmma_gemm_kernel<<<grid, blk128, 0, stream>>>(h, Wqkv, bqkv, nullptr, big,
                                                  ROWS_, 3 * DM_, DM_);
  }
  {
    dim3 grid(L_ / 64, B_ * AH_);
    flash_attn_kernel<<<grid, blk128, 0, stream>>>(big, yb);
  }
  {  // x2 <- x2 + o @ Wo + bo  (element-wise in-place residual is safe)
    dim3 grid(DM_ / 64, ROWS_ / 64);
    wmma_gemm_kernel<<<grid, blk128, 0, stream>>>(yb, Wo, bo, x2, x2,
                                                  ROWS_, DM_, DM_);
  }

  // ---- FFN branch ----
  ln512_kernel<<<ROWS_ / 8, blk256, 0, stream>>>(x2, g2, b2, h);
  {  // ff = h @ W1 + b1f
    dim3 grid(2048 / 64, ROWS_ / 64);
    wmma_gemm_kernel<<<grid, blk128, 0, stream>>>(h, W1, b1f, nullptr, big,
                                                  ROWS_, 2048, DM_);
  }
  gelu_mul_kernel<<<(ROWS_ * 256) / 256, blk256, 0, stream>>>(big, xbc);
  {  // out = x2 + g @ W2 + b2f
    dim3 grid(DM_ / 64, ROWS_ / 64);
    wmma_gemm_kernel<<<grid, blk128, 0, stream>>>(xbc, W2, b2f, x2, out,
                                                  ROWS_, DM_, DIN_);
  }
}
